// WordLSTMCell_22995254903543
// MI455X (gfx1250) — matrix operand order repaired and checked by reading, not compiled, not run on hardware
//
#include <hip/hip_runtime.h>

typedef __attribute__((ext_vector_type(16))) __bf16 bf16x16;
typedef __attribute__((ext_vector_type(2)))  __bf16 bf16x2;
typedef __attribute__((ext_vector_type(8)))  float  floatx8;

constexpr int IN_SZ  = 1024;
constexpr int HID    = 2048;
constexpr int NGATE  = 3 * HID;       // 6144
constexpr int BATCH  = 8192;
constexpr int M_TILE = 128;
constexpr int N_TILE = 64;
constexpr int KC     = 32;
constexpr int NKX    = IN_SZ / KC;    // 32 k-steps over x/W
constexpr int NK     = (IN_SZ + HID) / KC;   // 96 total
constexpr int LD     = 40;            // LDS row stride in bf16 elems (80 B: 16B-aligned, bank-spread)

#if defined(__has_builtin)
#if __has_builtin(__builtin_amdgcn_cvt_pk_bf16_f32)
#define HAVE_CVT_PK_BF16 1
#endif
#endif

// pack two f32 into one dword of 2x bf16 (lo, hi) -- single v_cvt_pk_bf16_f32
__device__ __forceinline__ unsigned pk_bf16(float lo, float hi) {
#ifdef HAVE_CVT_PK_BF16
  union { bf16x2 v; unsigned u; } c;
  c.v = __builtin_amdgcn_cvt_pk_bf16_f32(lo, hi);
  return c.u;
#else
  unsigned r;
  asm("v_cvt_pk_bf16_f32 %0, %1, %2" : "=v"(r) : "v"(lo), "v"(hi));
  return r;
#endif
}

__device__ __forceinline__ float fast_sigmoid(float v) {
  const float e = __builtin_amdgcn_exp2f(-1.44269504f * v);   // v_exp_f32
  return __builtin_amdgcn_rcpf(1.0f + e);                     // v_rcp_f32
}

union FragBF { uint4 u[2]; bf16x16 v; };

__global__ __launch_bounds__(256) void lstm_cell_wmma(
    const float* __restrict__ x,  const float* __restrict__ h0,
    const float* __restrict__ c0, const float* __restrict__ W,
    const float* __restrict__ Wh, const float* __restrict__ bias,
    float* __restrict__ out)
{
  __shared__ unsigned short As[M_TILE * LD];       // 10 KB (A tile, bf16, K-contiguous rows)
  __shared__ unsigned short Bs[3 * N_TILE * LD];   // 15 KB (3 gate B tiles, transposed: n-major, K-contiguous)

  const int tid     = threadIdx.x;
  const int lane    = tid & 31;
  const int wave    = tid >> 5;
  const int colBase = blockIdx.x * N_TILE;   // over HID (2048)
  const int rowBase = blockIdx.y * M_TILE;   // over BATCH (8192)

  const int wn     = (wave & 3) * 16;        // wave's n-subtile offset
  const int wmBase = (wave >> 2) * 16;       // wave covers m-subtiles wmBase + {0,32,64,96}

  // global-loader coords (256 threads)
  const int aq16 = (tid & 1) * 16;  // half-row offset: each thread loads 16 consecutive floats
  const int arr  = tid >> 1;        // A row 0..127 (single pass)
  const int bn4  = (tid & 15) * 4;  // B column group (4 consecutive n)
  const int bkp  = tid >> 4;        // B k-pair index 0..15 -> k = 2*bkp, 2*bkp+1

  // per-thread base pointers for the two K phases (x@W then h@Wh);
  // per-iteration offset is a uniform scalar, unrolled deltas are small IOFFSET constants
  const float* axBase = x  + (size_t)(rowBase + arr) * IN_SZ + aq16;
  const float* ahBase = h0 + (size_t)(rowBase + arr) * HID  + aq16;
  const float* bxBase = W  + (size_t)(2 * bkp) * NGATE + colBase + bn4;
  const float* bhBase = Wh + (size_t)(2 * bkp) * NGATE + colBase + bn4;

  // WMMA fragment coords (contiguous K order, same for A and B: lane half h holds K 16h..16h+15)
  const int lr  = lane & 15;
  const int kbA = (lane >> 4) * 16;   // lanes 0-15: K 0-15       ; lanes 16-31: K 16-31
  const int kbB = (lane >> 4) * 16;   // lanes 0-15: K 0-15       ; lanes 16-31: K 16-31

  floatx8 acc[4][3] = {};             // [m-subtile][gate f/i/c]

  float4 aReg[4];                     // 16 consecutive floats of one A row
  float4 bReg[3][2];                  // [gate][even-k / odd-k row]

  auto global_fetch = [&](int kt) {
    const bool xphase = (kt < NKX);
    const int  kk     = xphase ? kt : (kt - NKX);
    const float* ap = (xphase ? axBase : ahBase) + kk * KC;
    const float* bp = (xphase ? bxBase : bhBase) + (size_t)kk * (KC * NGATE);
#pragma unroll
    for (int q = 0; q < 4; ++q)
      aReg[q] = *(const float4*)(ap + q * 4);                // one base, 0/16/32/48 B offsets
#pragma unroll
    for (int g = 0; g < 3; ++g) {
      bReg[g][0] = *(const float4*)(bp + g * HID);           // 0/8K/16K offsets
      bReg[g][1] = *(const float4*)(bp + g * HID + NGATE);   // +24K offsets
    }
  };

  global_fetch(0);

  for (int kt = 0; kt < NK; ++kt) {
    // stage registers -> LDS with packed f32 -> bf16 convert (16 elems -> 2x b128 stores)
    {
      uint4 p0, p1;
      p0.x = pk_bf16(aReg[0].x, aReg[0].y);
      p0.y = pk_bf16(aReg[0].z, aReg[0].w);
      p0.z = pk_bf16(aReg[1].x, aReg[1].y);
      p0.w = pk_bf16(aReg[1].z, aReg[1].w);
      p1.x = pk_bf16(aReg[2].x, aReg[2].y);
      p1.y = pk_bf16(aReg[2].z, aReg[2].w);
      p1.z = pk_bf16(aReg[3].x, aReg[3].y);
      p1.w = pk_bf16(aReg[3].z, aReg[3].w);
      *(uint4*)&As[arr * LD + aq16]     = p0;
      *(uint4*)&As[arr * LD + aq16 + 8] = p1;
    }
#pragma unroll
    for (int g = 0; g < 3; ++g) {
      // transpose: one packed (k even, k odd) dword per output column n
      const unsigned u0 = pk_bf16(bReg[g][0].x, bReg[g][1].x);
      const unsigned u1 = pk_bf16(bReg[g][0].y, bReg[g][1].y);
      const unsigned u2 = pk_bf16(bReg[g][0].z, bReg[g][1].z);
      const unsigned u3 = pk_bf16(bReg[g][0].w, bReg[g][1].w);
      *(unsigned*)&Bs[(g * N_TILE + bn4 + 0) * LD + 2 * bkp] = u0;
      *(unsigned*)&Bs[(g * N_TILE + bn4 + 1) * LD + 2 * bkp] = u1;
      *(unsigned*)&Bs[(g * N_TILE + bn4 + 2) * LD + 2 * bkp] = u2;
      *(unsigned*)&Bs[(g * N_TILE + bn4 + 3) * LD + 2 * bkp] = u3;
    }
    __syncthreads();

    if (kt + 1 < NK) global_fetch(kt + 1);   // overlap next tile's VMEM with WMMA below

    FragBF a[4], b[3];
#pragma unroll
    for (int mi = 0; mi < 4; ++mi) {
      const int m = wmBase + mi * 32 + lr;
      a[mi].u[0] = *(const uint4*)&As[m * LD + kbA];        // elems 0-7  : K kbA..kbA+7
      a[mi].u[1] = *(const uint4*)&As[m * LD + kbA + 8];    // elems 8-15 : K kbA+8..
    }
#pragma unroll
    for (int g = 0; g < 3; ++g) {
      const int n = wn + lr;
      b[g].u[0] = *(const uint4*)&Bs[(g * N_TILE + n) * LD + kbB];
      b[g].u[1] = *(const uint4*)&Bs[(g * N_TILE + n) * LD + kbB + 8];
    }
#pragma unroll
    for (int mi = 0; mi < 4; ++mi)
#pragma unroll
      for (int g = 0; g < 3; ++g)
        acc[mi][g] = __builtin_amdgcn_wmma_f32_16x16x32_bf16(
            false, a[mi].v, false, b[g].v, (short)0, acc[mi][g], false, false);
    __syncthreads();
  }

  // epilogue: bias + sigmoid/tanh cell math, fused with c0 read and c1 store
  const int col = colBase + wn + lr;
  const float bF = bias[col];
  const float bI = bias[col + HID];
  const float bC = bias[col + 2 * HID];
  const int mh = (lane >> 4) * 8;   // C/D layout: lanes 16-31 hold M = r + 8

#pragma unroll
  for (int mi = 0; mi < 4; ++mi)
#pragma unroll
    for (int r = 0; r < 8; ++r) {
      const int m = wmBase + mi * 32 + mh + r;
      const size_t idx = (size_t)(rowBase + m) * HID + col;
      const float f  = acc[mi][0][r] + bF;
      const float i  = acc[mi][1][r] + bI;
      const float cc = acc[mi][2][r] + bC;
      const float sf = fast_sigmoid(f);
      const float si = fast_sigmoid(i);
      const float th = 2.0f * fast_sigmoid(2.0f * cc) - 1.0f;   // tanh(c)
      out[idx] = fmaf(sf, c0[idx], si * th);
    }
}

extern "C" void kernel_launch(void* const* d_in, const int* in_sizes, int n_in,
                              void* d_out, int out_size, void* d_ws, size_t ws_size,
                              hipStream_t stream) {
  const float* x    = (const float*)d_in[0];
  const float* h0   = (const float*)d_in[1];
  const float* c0   = (const float*)d_in[2];
  const float* W    = (const float*)d_in[3];
  const float* Wh   = (const float*)d_in[4];
  const float* bias = (const float*)d_in[5];
  dim3 grid(HID / N_TILE, BATCH / M_TILE);   // (32, 64)
  lstm_cell_wmma<<<grid, 256, 0, stream>>>(x, h0, c0, W, Wh, bias, (float*)d_out);
}
